// MultiheadAttention820_46480136077896
// MI455X (gfx1250) — matrix operand order repaired and checked by reading, not compiled, not run on hardware
//
#include <hip/hip_runtime.h>

typedef __bf16 bf16_t;
typedef __attribute__((ext_vector_type(16))) __bf16 v16bf;
typedef __attribute__((ext_vector_type(8)))  __bf16 v8bf;
typedef __attribute__((ext_vector_type(8)))  float  v8f;

#define T_LEN 2048
#define B_SZ  4
#define E_DIM 1024
#define H_N   16
#define D_HD  64
#define K_TAPS 15
#define PAD_LEFT 7
#define M_ROWS (T_LEN * B_SZ)   /* 8192 */

// ---------------------------------------------------------------------------
// WMMA fragment loaders (wave32, 16x16x32 bf16).
// A fragment (16xK=32, row-major, leading dim ld):
//   lane L holds row M = L%16; elements i: K = (i&7) + 8*(L/16) + 16*(i/8)
//   -> two contiguous 8-element (16B) chunks at kbase and kbase+16.
// B fragment (K=32 x 16 from B^T row-major [N][K]):
//   lane L holds col N = L%16; elements i: K = (i&7) + 8*(L/16) + 16*(i/8)
//   -> two contiguous 8-element (16B) chunks at kbase and kbase+16.
// C/D fragment: VGPR j, lane L -> (M = j + 8*(L/16), N = L%16).
// ---------------------------------------------------------------------------
__device__ __forceinline__ v16bf load_afrag(const bf16_t* base, int ld, int lane) {
  const bf16_t* p = base + (size_t)(lane & 15) * ld + ((lane >> 4) << 3);
  v8bf lo = *reinterpret_cast<const v8bf*>(p);
  v8bf hi = *reinterpret_cast<const v8bf*>(p + 16);
  v16bf f;
#pragma unroll
  for (int i = 0; i < 8; ++i) { f[i] = lo[i]; f[i + 8] = hi[i]; }
  return f;
}

__device__ __forceinline__ v16bf load_bfrag(const bf16_t* base, int ld, int lane) {
  const bf16_t* p = base + (size_t)(lane & 15) * ld + ((lane >> 4) << 3);
  v8bf lo = *reinterpret_cast<const v8bf*>(p);
  v8bf hi = *reinterpret_cast<const v8bf*>(p + 16);
  v16bf f;
#pragma unroll
  for (int i = 0; i < 8; ++i) { f[i] = lo[i]; f[i + 8] = hi[i]; }
  return f;
}

__device__ __forceinline__ v8f wmma_bf16(v16bf a, v16bf b, v8f c) {
  return __builtin_amdgcn_wmma_f32_16x16x32_bf16(false, a, false, b, (short)0, c,
                                                 false, false);
}

// ---------------------------------------------------------------------------
// f32 -> bf16 conversion
// ---------------------------------------------------------------------------
__global__ void cvt_f32_to_bf16(const float* __restrict__ src,
                                bf16_t* __restrict__ dst, int n) {
  int i = blockIdx.x * blockDim.x + threadIdx.x;
  if (i < n) dst[i] = (bf16_t)src[i];
}

// ---------------------------------------------------------------------------
// QKV GEMM (M=8192, N=3072, K=1024), 4 waves/block, 16x64 per wave.
// All 5 fragments of a k-step are loaded up front (distinct live values) so
// the scheduler can clause the loads and overlap them with the wmma batch.
// Packing epilogue:
//   q -> Qh[b*16+h][t][d] * 0.125   (bf16)
//   k -> Kh[b*16+h][t][d]           (bf16)
//   v -> Vbf[m][c]  and  VhT[b*16+h][d][t]  (bf16)
// ---------------------------------------------------------------------------
__global__ __launch_bounds__(128) void qkv_gemm_pack(
    const bf16_t* __restrict__ xbf, const bf16_t* __restrict__ win,
    const float* __restrict__ in_b, bf16_t* __restrict__ qh,
    bf16_t* __restrict__ kh, bf16_t* __restrict__ vbf,
    bf16_t* __restrict__ vt) {
  int lane = threadIdx.x & 31;
  int wave = threadIdx.x >> 5;
  int m0 = (blockIdx.x * 4 + wave) << 4;
  int n0 = blockIdx.y << 6;                 // 64-wide N strip
  v8f acc0 = {}, acc1 = {}, acc2 = {}, acc3 = {};
  const bf16_t* arow = xbf + (size_t)m0 * E_DIM;
  const bf16_t* brow = win + (size_t)n0 * E_DIM;
#pragma unroll 2
  for (int k0 = 0; k0 < E_DIM; k0 += 32) {
    v16bf a  = load_afrag(arow + k0, E_DIM, lane);
    v16bf b0 = load_bfrag(brow + k0, E_DIM, lane);
    v16bf b1 = load_bfrag(brow + (size_t)16 * E_DIM + k0, E_DIM, lane);
    v16bf b2 = load_bfrag(brow + (size_t)32 * E_DIM + k0, E_DIM, lane);
    v16bf b3 = load_bfrag(brow + (size_t)48 * E_DIM + k0, E_DIM, lane);
    acc0 = wmma_bf16(a, b0, acc0);
    acc1 = wmma_bf16(a, b1, acc1);
    acc2 = wmma_bf16(a, b2, acc2);
    acc3 = wmma_bf16(a, b3, acc3);
  }
  int moff = (lane >> 4) << 3;
  v8f accs[4] = {acc0, acc1, acc2, acc3};
#pragma unroll
  for (int i = 0; i < 4; ++i) {
    int col = n0 + i * 16 + (lane & 15);
    float bi = in_b[col];
#pragma unroll
    for (int j = 0; j < 8; ++j) {
      int m = m0 + j + moff;
      int t = m >> 2, bb = m & 3;
      float v = accs[i][j] + bi;
      if (col < E_DIM) {
        int h = col >> 6, d = col & 63;
        qh[(((size_t)(bb * H_N + h)) * T_LEN + t) * D_HD + d] =
            (bf16_t)(v * 0.125f);
      } else if (col < 2 * E_DIM) {
        int c = col - E_DIM;
        int h = c >> 6, d = c & 63;
        kh[(((size_t)(bb * H_N + h)) * T_LEN + t) * D_HD + d] = (bf16_t)v;
      } else {
        int c = col - 2 * E_DIM;
        int h = c >> 6, d = c & 63;
        vbf[(size_t)m * E_DIM + c] = (bf16_t)v;
        vt[(((size_t)(bb * H_N + h)) * D_HD + d) * T_LEN + t] = (bf16_t)v;
      }
    }
  }
}

// ---------------------------------------------------------------------------
// C[M,N] = A[M,K] @ Bt[N,K]^T + bias, f32 out, 4 waves/block, 16x64 per wave.
// Requires N % 64 == 0.
// ---------------------------------------------------------------------------
__global__ __launch_bounds__(128) void gemm_bt_bias_f32_n4(
    const bf16_t* __restrict__ A, const bf16_t* __restrict__ Bt,
    const float* __restrict__ bias, float* __restrict__ C, int N, int Kd) {
  int lane = threadIdx.x & 31;
  int wave = threadIdx.x >> 5;
  int m0 = (blockIdx.x * 4 + wave) << 4;
  int n0 = blockIdx.y << 6;
  v8f acc0 = {}, acc1 = {}, acc2 = {}, acc3 = {};
  const bf16_t* arow = A + (size_t)m0 * Kd;
  const bf16_t* brow = Bt + (size_t)n0 * Kd;
#pragma unroll 2
  for (int k0 = 0; k0 < Kd; k0 += 32) {
    v16bf a  = load_afrag(arow + k0, Kd, lane);
    v16bf b0 = load_bfrag(brow + k0, Kd, lane);
    v16bf b1 = load_bfrag(brow + (size_t)16 * Kd + k0, Kd, lane);
    v16bf b2 = load_bfrag(brow + (size_t)32 * Kd + k0, Kd, lane);
    v16bf b3 = load_bfrag(brow + (size_t)48 * Kd + k0, Kd, lane);
    acc0 = wmma_bf16(a, b0, acc0);
    acc1 = wmma_bf16(a, b1, acc1);
    acc2 = wmma_bf16(a, b2, acc2);
    acc3 = wmma_bf16(a, b3, acc3);
  }
  int moff = (lane >> 4) << 3;
  v8f accs[4] = {acc0, acc1, acc2, acc3};
#pragma unroll
  for (int i = 0; i < 4; ++i) {
    int col = n0 + i * 16 + (lane & 15);
    float bi = bias[col];
#pragma unroll
    for (int j = 0; j < 8; ++j) {
      int m = m0 + j + moff;
      C[(size_t)m * N + col] = accs[i][j] + bi;
    }
  }
}

// ---------------------------------------------------------------------------
// C[M,N] = A[M,K] @ Bt[N,K]^T + bias, f32 out, 1 wave = one 16x16 tile
// (used for N=240 dyn-conv logits, not divisible by 64).
// ---------------------------------------------------------------------------
__global__ __launch_bounds__(32) void gemm_bt_bias_f32(
    const bf16_t* __restrict__ A, const bf16_t* __restrict__ Bt,
    const float* __restrict__ bias, float* __restrict__ C, int N, int Kd) {
  int lane = threadIdx.x;
  int m0 = blockIdx.x << 4;
  int n0 = blockIdx.y << 4;
  v8f acc = {};
  const bf16_t* arow = A + (size_t)m0 * Kd;
  const bf16_t* brow = Bt + (size_t)n0 * Kd;
#pragma unroll 4
  for (int k0 = 0; k0 < Kd; k0 += 32) {
    v16bf a = load_afrag(arow + k0, Kd, lane);
    v16bf b = load_bfrag(brow + k0, Kd, lane);
    acc = wmma_bf16(a, b, acc);
  }
  int col = n0 + (lane & 15);
  float bi = bias[col];
  int moff = (lane >> 4) << 3;
#pragma unroll
  for (int j = 0; j < 8; ++j) {
    int m = m0 + j + moff;
    C[(size_t)m * N + col] = acc[j] + bi;
  }
}

// ---------------------------------------------------------------------------
// Flash attention, one wave per (head, 16-query tile). Writes cat[:, 0:E].
// ---------------------------------------------------------------------------
__global__ __launch_bounds__(32) void flash_attn(
    const bf16_t* __restrict__ qh, const bf16_t* __restrict__ kh,
    const bf16_t* __restrict__ vt, bf16_t* __restrict__ cat) {
  __shared__ bf16_t ldsP[16 * 32];
  int lane = threadIdx.x;
  int qt = blockIdx.x;   // 0..127
  int head = blockIdx.y; // 0..63
  const bf16_t* Q  = qh + ((size_t)head * T_LEN + (size_t)qt * 16) * D_HD;
  const bf16_t* Kb = kh + (size_t)head * T_LEN * D_HD;
  const bf16_t* VT = vt + (size_t)head * D_HD * T_LEN;

  v16bf qf0 = load_afrag(Q, D_HD, lane);
  v16bf qf1 = load_afrag(Q + 32, D_HD, lane);

  v8f o0 = {}, o1 = {}, o2 = {}, o3 = {};
  float rmax[8], lsum[8];
#pragma unroll
  for (int j = 0; j < 8; ++j) { rmax[j] = -1e30f; lsum[j] = 0.f; }

  for (int s0 = 0; s0 < T_LEN; s0 += 32) {
    // S tile: 16 queries x 32 keys (two 16x16 C frags), K-dim = D = 64
    v8f sa = {}, sb = {};
    {
      const bf16_t* k0p = Kb + (size_t)s0 * D_HD;
      const bf16_t* k1p = Kb + (size_t)(s0 + 16) * D_HD;
      v16bf b0 = load_bfrag(k0p, D_HD, lane);
      v16bf b1 = load_bfrag(k0p + 32, D_HD, lane);
      v16bf b2 = load_bfrag(k1p, D_HD, lane);
      v16bf b3 = load_bfrag(k1p + 32, D_HD, lane);
      sa = wmma_bf16(qf0, b0, sa);
      sa = wmma_bf16(qf1, b1, sa);
      sb = wmma_bf16(qf0, b2, sb);
      sb = wmma_bf16(qf1, b3, sb);
    }
    // online softmax (row reductions across the 16-lane half-wave)
#pragma unroll
    for (int j = 0; j < 8; ++j) {
      float mv = fmaxf(sa[j], sb[j]);
      mv = fmaxf(mv, __shfl_xor(mv, 1, 32));
      mv = fmaxf(mv, __shfl_xor(mv, 2, 32));
      mv = fmaxf(mv, __shfl_xor(mv, 4, 32));
      mv = fmaxf(mv, __shfl_xor(mv, 8, 32));
      float newm = fmaxf(rmax[j], mv);
      float alpha = __expf(rmax[j] - newm);
      rmax[j] = newm;
      float p0 = __expf(sa[j] - newm);
      float p1 = __expf(sb[j] - newm);
      sa[j] = p0; sb[j] = p1;
      float rs = p0 + p1;
      rs += __shfl_xor(rs, 1, 32);
      rs += __shfl_xor(rs, 2, 32);
      rs += __shfl_xor(rs, 4, 32);
      rs += __shfl_xor(rs, 8, 32);
      lsum[j] = lsum[j] * alpha + rs;
      o0[j] *= alpha; o1[j] *= alpha; o2[j] *= alpha; o3[j] *= alpha;
    }
    // C-fragment -> A-fragment relayout for P via LDS (16x32 bf16 row-major)
    {
      int nlo = lane & 15;
      int mhi = (lane >> 4) << 3;
#pragma unroll
      for (int j = 0; j < 8; ++j) {
        ldsP[(j + mhi) * 32 + nlo]      = (bf16_t)sa[j];
        ldsP[(j + mhi) * 32 + nlo + 16] = (bf16_t)sb[j];
      }
    }
    __syncthreads();
    v16bf pf = load_afrag(ldsP, 32, lane);
    {
      const bf16_t* vb = VT + s0;
      v16bf v0 = load_bfrag(vb, T_LEN, lane);
      v16bf v1 = load_bfrag(vb + (size_t)16 * T_LEN, T_LEN, lane);
      v16bf v2 = load_bfrag(vb + (size_t)32 * T_LEN, T_LEN, lane);
      v16bf v3 = load_bfrag(vb + (size_t)48 * T_LEN, T_LEN, lane);
      o0 = wmma_bf16(pf, v0, o0);
      o1 = wmma_bf16(pf, v1, o1);
      o2 = wmma_bf16(pf, v2, o2);
      o3 = wmma_bf16(pf, v3, o3);
    }
    __syncthreads();
  }
  // epilogue: normalize and scatter into cat[:, 0:E]
  int bb = head >> 4, h = head & 15;
  int nlo = lane & 15;
  int mhi = (lane >> 4) << 3;
#pragma unroll
  for (int j = 0; j < 8; ++j) {
    float inv = 1.f / lsum[j];
    int t = qt * 16 + j + mhi;
    size_t row = (size_t)(t * B_SZ + bb) * (2 * E_DIM) + (size_t)h * D_HD + nlo;
    cat[row]      = (bf16_t)(o0[j] * inv);
    cat[row + 16] = (bf16_t)(o1[j] * inv);
    cat[row + 32] = (bf16_t)(o2[j] * inv);
    cat[row + 48] = (bf16_t)(o3[j] * inv);
  }
}

// ---------------------------------------------------------------------------
// Dynamic conv: softmax over K=15 taps + depthwise conv; writes cat[:, E:2E].
// ---------------------------------------------------------------------------
__global__ void dynconv_kernel(const float* __restrict__ wlog,
                               const bf16_t* __restrict__ vbf,
                               bf16_t* __restrict__ cat) {
  int idx = blockIdx.x * blockDim.x + threadIdx.x;
  if (idx >= M_ROWS * E_DIM) return;
  int m = idx >> 10;
  int c = idx & 1023;
  int h = c >> 6;
  int t = m >> 2, bb = m & 3;
  const float* lg = wlog + (size_t)m * (H_N * K_TAPS) + h * K_TAPS;
  float mx = -1e30f;
#pragma unroll
  for (int k = 0; k < K_TAPS; ++k) mx = fmaxf(mx, lg[k]);
  float w[K_TAPS];
  float sum = 0.f;
#pragma unroll
  for (int k = 0; k < K_TAPS; ++k) { w[k] = __expf(lg[k] - mx); sum += w[k]; }
  float acc = 0.f;
#pragma unroll
  for (int k = 0; k < K_TAPS; ++k) {
    int tt = t + k - PAD_LEFT;
    if (tt >= 0 && tt < T_LEN)
      acc += w[k] * (float)vbf[(size_t)(tt * B_SZ + bb) * E_DIM + c];
  }
  cat[(size_t)m * (2 * E_DIM) + E_DIM + c] = (bf16_t)(acc / sum);
}

// ---------------------------------------------------------------------------
// Workspace layout (bytes, 256B aligned)
// ---------------------------------------------------------------------------
#define OFF_XBF   0u
#define OFF_WIN   (OFF_XBF  + 16777216u)  /* x bf16: 8192*1024*2      */
#define OFF_WOUT  (OFF_WIN  + 6291456u)   /* in_proj_w bf16           */
#define OFF_WDYN  (OFF_WOUT + 4194304u)   /* out_proj_w bf16          */
#define OFF_QH    (OFF_WDYN + 491520u)    /* dyn_w bf16               */
#define OFF_KH    (OFF_QH   + 16777216u)
#define OFF_VT    (OFF_KH   + 16777216u)
#define OFF_VBF   (OFF_VT   + 16777216u)
#define OFF_WLOG  (OFF_VBF  + 16777216u)
#define OFF_CAT   (OFF_WLOG + 7864320u)   /* wlog f32: 8192*240*4     */
/* total = OFF_CAT + 33554432 = 136282112 bytes (~130 MB) */

extern "C" void kernel_launch(void* const* d_in, const int* in_sizes, int n_in,
                              void* d_out, int out_size, void* d_ws,
                              size_t ws_size, hipStream_t stream) {
  (void)in_sizes; (void)n_in; (void)out_size; (void)ws_size;
  const float* x     = (const float*)d_in[0];
  const float* in_w  = (const float*)d_in[1];
  const float* in_b  = (const float*)d_in[2];
  const float* out_w = (const float*)d_in[3];
  const float* out_b = (const float*)d_in[4];
  const float* dyn_w = (const float*)d_in[5];
  const float* dyn_b = (const float*)d_in[6];
  float* out = (float*)d_out;

  char* ws = (char*)d_ws;
  bf16_t* xbf  = (bf16_t*)(ws + OFF_XBF);
  bf16_t* win  = (bf16_t*)(ws + OFF_WIN);
  bf16_t* wout = (bf16_t*)(ws + OFF_WOUT);
  bf16_t* wdyn = (bf16_t*)(ws + OFF_WDYN);
  bf16_t* qh   = (bf16_t*)(ws + OFF_QH);
  bf16_t* kh   = (bf16_t*)(ws + OFF_KH);
  bf16_t* vt   = (bf16_t*)(ws + OFF_VT);
  bf16_t* vbf  = (bf16_t*)(ws + OFF_VBF);
  float*  wlog = (float*)(ws + OFF_WLOG);
  bf16_t* cat  = (bf16_t*)(ws + OFF_CAT);

  // 1) convert inputs/weights to bf16
  cvt_f32_to_bf16<<<(M_ROWS * E_DIM + 255) / 256, 256, 0, stream>>>(x, xbf, M_ROWS * E_DIM);
  cvt_f32_to_bf16<<<(3 * E_DIM * E_DIM + 255) / 256, 256, 0, stream>>>(in_w, win, 3 * E_DIM * E_DIM);
  cvt_f32_to_bf16<<<(E_DIM * 2 * E_DIM + 255) / 256, 256, 0, stream>>>(out_w, wout, E_DIM * 2 * E_DIM);
  cvt_f32_to_bf16<<<(H_N * K_TAPS * E_DIM + 255) / 256, 256, 0, stream>>>(dyn_w, wdyn, H_N * K_TAPS * E_DIM);

  // 2) fused QKV projection + head packing (WMMA, 16x64 per wave)
  qkv_gemm_pack<<<dim3(M_ROWS / 64, 3 * E_DIM / 64), 128, 0, stream>>>(
      xbf, win, in_b, qh, kh, vbf, vt);

  // 3) dynamic-conv weight logits GEMM (WMMA) + conv branch -> cat[:, E:2E]
  gemm_bt_bias_f32<<<dim3(M_ROWS / 16, (H_N * K_TAPS) / 16), 32, 0, stream>>>(
      vbf, wdyn, dyn_b, wlog, H_N * K_TAPS, E_DIM);
  dynconv_kernel<<<(M_ROWS * E_DIM) / 256, 256, 0, stream>>>(wlog, vbf, cat);

  // 4) flash attention (WMMA) -> cat[:, 0:E]
  flash_attn<<<dim3(T_LEN / 16, B_SZ * H_N), 32, 0, stream>>>(qh, kh, vt, cat);

  // 5) output projection (WMMA, 16x64 per wave) -> d_out (f32)
  gemm_bt_bias_f32_n4<<<dim3(M_ROWS / 64, E_DIM / 64), 128, 0, stream>>>(
      cat, wout, out_b, out, E_DIM, 2 * E_DIM);
}
